// Gemma4AudioLocalAttention_5978594476551
// MI455X (gfx1250) — matrix operand order repaired and checked by reading, not compiled, not run on hardware
//
#include <hip/hip_runtime.h>
#include <hip/hip_bf16.h>
#include <math.h>

#define HIDDEN   1536
#define HEADS    12
#define HEAD_DIM 128
#define CHUNKSZ  128
#define NCTX     255
#define NBLK     32
#define TSEQ     4096
#define BATCH    4

// padded LDS row strides (halves) to avoid bank conflicts on b128 fragment reads
#define BSTRIDE  40
#define KSTRIDE  136
#define VSTRIDE  40
#define PSTRIDE  40

typedef _Float16 half8_t  __attribute__((ext_vector_type(8)));
typedef _Float16 half16_t __attribute__((ext_vector_type(16)));
typedef float    float8_t __attribute__((ext_vector_type(8)));

union H16u { half16_t v; half8_t h8[2]; _Float16 e[16]; };

__device__ __forceinline__ int iclamp(int v, int lo, int hi) {
  return v < lo ? lo : (v > hi ? hi : v);
}

// WMMA A-fragment (16x32 f16, ISA 7.12.2): lane row = lid&15,
// halves [0..7] = K (hi16?8:0)+0..7, halves [8..15] = K 16+(hi16?8:0)+0..7
__device__ __forceinline__ half16_t load_a_frag(const _Float16* __restrict__ base,
                                                int row, long stride, int k0, int hi16) {
  const _Float16* p = base + (long)row * stride + k0 + hi16 * 8;
  H16u u;
  u.h8[0] = *(const half8_t*)(p);
  u.h8[1] = *(const half8_t*)(p + 16);
  return u.v;
}

// WMMA B-fragment (32x16 f16): lane column n = lid&15, halves [0..7] = K (hi16?8:0)+0..7, [8..15] = K 16+(hi16?8:0)+0..7.
// Source is N-major: element [n][k] contiguous in k.
__device__ __forceinline__ half16_t load_b_frag(const _Float16* __restrict__ base,
                                                int n_row, long stride, int k0, int hi16) {
  const _Float16* p = base + (long)n_row * stride + k0 + hi16 * 8;
  H16u u;
  u.h8[0] = *(const half8_t*)(p);
  u.h8[1] = *(const half8_t*)(p + 16);
  return u.v;
}

// ---------------- prep kernels ----------------

__global__ void conv_x_kernel(const float* __restrict__ src, _Float16* __restrict__ dst, long n) {
  long i = (long)blockIdx.x * blockDim.x + threadIdx.x;
  if (i < n) dst[i] = (_Float16)src[i];
}

__global__ void transpose_w_kernel(const float* __restrict__ src, _Float16* __restrict__ dst) {
  long i = (long)blockIdx.x * blockDim.x + threadIdx.x;
  if (i < (long)HIDDEN * HIDDEN) {
    int k = (int)(i / HIDDEN), n = (int)(i % HIDDEN);
    dst[(long)n * HIDDEN + k] = (_Float16)src[i];  // dst[n][k] = W[k][n]
  }
}

__global__ void sinemb_kernel(_Float16* __restrict__ dst) {
  int i = blockIdx.x * blockDim.x + threadIdx.x;
  if (i < 128 * HIDDEN) {
    int p = i / HIDDEN, c = i % HIDDEN;
    int ts = (c < 768) ? c : (c - 768);
    float inv = expf(-(float)ts * (logf(10000.0f) / 767.0f));
    float arg = (float)(127 - p) * inv;     // positions = 127,126,...,0
    float v = (c < 768) ? sinf(arg) : cosf(arg);
    dst[i] = (_Float16)v;
  }
}

// ---------------- f16 WMMA GEMM: out[M,1536] = A[M,1536] @ W, W given N-major ----------------
// LDS double-buffered B tile: 8x cut in L2 traffic vs per-wave fragment streaming.
// mode 0: q (per-dim softplus scale), 1: k (K_SCALE), 2: v (store transposed [b,h,d,t]), 3: pos
__global__ __launch_bounds__(256) void gemm_f16_kernel(
    const _Float16* __restrict__ A, const _Float16* __restrict__ Bt,
    _Float16* __restrict__ outp, const float* __restrict__ pds, int M, int mode)
{
  const float QS = 0.08838834764831845f / 0.6931471805599453f;  // d^-0.5 / ln2
  const float KS = 1.3132616875182228f / 0.6931471805599453f;   // log1p(e) / ln2
  const int NKS = HIDDEN / 32;                                  // 48 k-steps

  __shared__ __align__(16) _Float16 Bst[2][128][BSTRIDE];       // 20 KB

  int mt = blockIdx.x / (HIDDEN / 128);
  int nt = blockIdx.x % (HIDDEN / 128);
  int w = threadIdx.x >> 5;
  int lid = threadIdx.x & 31;
  int hi16 = lid >> 4, l16 = lid & 15;
  int m0 = mt * 128 + w * 16;

  // cooperative staging role: 2 threads per n-row, 16 halves each
  int n_l  = threadIdx.x >> 1;
  int koff = (threadIdx.x & 1) * 16;
  const _Float16* wsrc = Bt + (long)(nt * 128 + n_l) * HIDDEN + koff;

  float8_t zz = {};
  float8_t acc[8];
#pragma unroll
  for (int j = 0; j < 8; ++j) acc[j] = zz;

  // prologue: stage k-step 0
  {
    H16u r;
    r.h8[0] = *(const half8_t*)(wsrc);
    r.h8[1] = *(const half8_t*)(wsrc + 8);
    *(half8_t*)&Bst[0][n_l][koff]     = r.h8[0];
    *(half8_t*)&Bst[0][n_l][koff + 8] = r.h8[1];
  }

  for (int ks = 0; ks < NKS; ++ks) {
    __syncthreads();                       // tile ks ready; prior reads of (ks+1)&1 retired
    H16u r;
    if (ks + 1 < NKS) {                    // prefetch next tile (overlaps WMMA below)
      const _Float16* p = wsrc + (ks + 1) * 32;
      r.h8[0] = *(const half8_t*)(p);
      r.h8[1] = *(const half8_t*)(p + 8);
    }
    half16_t a = load_a_frag(A, m0 + l16, HIDDEN, ks * 32, hi16);
    const _Float16* bb = &Bst[ks & 1][0][0];
#pragma unroll
    for (int j = 0; j < 8; ++j) {
      half16_t bf = load_b_frag(bb, j * 16 + l16, BSTRIDE, 0, hi16);
      acc[j] = __builtin_amdgcn_wmma_f32_16x16x32_f16(false, a, false, bf,
                                                      (short)0, acc[j], false, false);
    }
    if (ks + 1 < NKS) {                    // store into the other buffer (no conflict)
      *(half8_t*)&Bst[(ks + 1) & 1][n_l][koff]     = r.h8[0];
      *(half8_t*)&Bst[(ks + 1) & 1][n_l][koff + 8] = r.h8[1];
    }
  }

#pragma unroll
  for (int j = 0; j < 8; ++j) {
    int n = nt * 128 + j * 16 + l16;
    float scl = 1.0f;
    if (mode == 0) scl = QS * log1pf(expf(pds[n & 127]));
    else if (mode == 1) scl = KS;
#pragma unroll
    for (int r = 0; r < 8; ++r) {
      int m = m0 + hi16 * 8 + r;
      float val = acc[j][r] * scl;
      if (mode == 2) {   // v transposed: vt[((b*H + h)*128 + d)*T + t]
        int bb2 = m >> 12, t = m & (TSEQ - 1);
        int hh = n >> 7, d = n & 127;
        outp[((((long)bb2 * HEADS + hh) * HEAD_DIM + d) << 12) + t] = (_Float16)val;
      } else {
        outp[(long)m * HIDDEN + n] = (_Float16)val;
      }
    }
  }
}

// ---------------- fused local attention (flash-style, per (b,nb,h)) ----------------
__global__ __launch_bounds__(256) void attn_kernel(
    const _Float16* __restrict__ qh, const _Float16* __restrict__ kh,
    const _Float16* __restrict__ vt, const _Float16* __restrict__ posh,
    const unsigned char* __restrict__ mask, float* __restrict__ out)
{
  __shared__ __align__(16) _Float16 Tbd[8][16][128];     // rel-pos term per wave, 32 KB
  __shared__ __align__(16) _Float16 Pst[8][16][PSTRIDE]; // P tile staging, 10 KB
  __shared__ __align__(16) _Float16 Kst[32][KSTRIDE];    // shared K chunk, 8.5 KB
  __shared__ __align__(16) _Float16 Vst[128][VSTRIDE];   // shared V^T chunk, 10 KB

  int b = blockIdx.x / (NBLK * HEADS);
  int rem = blockIdx.x % (NBLK * HEADS);
  int nb = rem / HEADS;
  int h = rem % HEADS;
  int w = threadIdx.x >> 5;
  int lid = threadIdx.x & 31;
  int hi16 = lid >> 4, l16 = lid & 15;
  int i0 = w * 16;                               // query-row tile base within chunk

  // hoist Q A-fragments (K=128 -> 4 steps)
  int qrow = b * TSEQ + nb * CHUNKSZ + i0;
  half16_t aq[4];
#pragma unroll
  for (int ks = 0; ks < 4; ++ks)
    aq[ks] = load_a_frag(qh, qrow + l16, HIDDEN, h * HEAD_DIM + ks * 32, hi16);

  // Phase 1: Tbd[i][p] = q[i] . pos[p]   (16 x 128 via WMMA)
#pragma unroll
  for (int pt = 0; pt < 8; ++pt) {
    float8_t c = {};
#pragma unroll
    for (int ks = 0; ks < 4; ++ks) {
      half16_t bp = load_b_frag(posh, pt * 16 + l16, HIDDEN, h * HEAD_DIM + ks * 32, hi16);
      c = __builtin_amdgcn_wmma_f32_16x16x32_f16(false, aq[ks], false, bp,
                                                 (short)0, c, false, false);
    }
#pragma unroll
    for (int r = 0; r < 8; ++r)
      Tbd[w][hi16 * 8 + r][pt * 16 + l16] = (_Float16)c[r];
  }

  // Phase 2: flash attention over 8 chunks of 32 context columns
  float8_t zz = {};
  float8_t o[8];
#pragma unroll
  for (int dt = 0; dt < 8; ++dt) o[dt] = zz;
  float rmax[8], rsum[8];
#pragma unroll
  for (int r = 0; r < 8; ++r) { rmax[r] = -3.0e38f; rsum[r] = 0.0f; }

  const long kbbase = (long)b * TSEQ * HIDDEN + h * HEAD_DIM;
  const long vbase  = ((long)b * HEADS + h) * HEAD_DIM * TSEQ;
  const int toff = nb * CHUNKSZ - 127;  // abs time of context col 0

  for (int ck = 0; ck < 8; ++ck) {
    int j0 = ck * 32;

    __syncthreads();   // previous chunk's Kst/Vst reads retired in all waves
    {  // cooperative K chunk stage: rows j0..j0+31, 32B per thread
      int jl = threadIdx.x >> 3;
      int kseg = (threadIdx.x & 7) * 16;
      int t = iclamp(toff + j0 + jl, 0, TSEQ - 1);
      const _Float16* src = kh + kbbase + (long)t * HIDDEN + kseg;
      *(half8_t*)&Kst[jl][kseg]     = *(const half8_t*)(src);
      *(half8_t*)&Kst[jl][kseg + 8] = *(const half8_t*)(src + 8);
    }
    {  // cooperative V^T chunk stage: Vst[d][0..31] = vt[d][t00..t00+31] (odd offset -> memcpy)
      int d = threadIdx.x >> 1;
      int tseg = (threadIdx.x & 1) * 16;
      const _Float16* vrow = vt + vbase + (long)d * TSEQ;
      int t00 = toff + j0 + tseg;
      H16u u;
      if (t00 >= 0 && t00 + 15 < TSEQ) {
        __builtin_memcpy(&u, vrow + t00, 32);
      } else {
#pragma unroll
        for (int i2 = 0; i2 < 16; ++i2)
          u.e[i2] = vrow[iclamp(t00 + i2, 0, TSEQ - 1)];
      }
      *(half8_t*)&Vst[d][tseg]     = u.h8[0];
      *(half8_t*)&Vst[d][tseg + 8] = u.h8[1];
    }
    __syncthreads();   // chunk staged

    float8_t s0 = zz, s1 = zz;
#pragma unroll
    for (int ks = 0; ks < 4; ++ks) {
      half16_t b0 = load_b_frag(&Kst[0][0], l16,      KSTRIDE, ks * 32, hi16);
      half16_t b1 = load_b_frag(&Kst[0][0], 16 + l16, KSTRIDE, ks * 32, hi16);
      s0 = __builtin_amdgcn_wmma_f32_16x16x32_f16(false, aq[ks], false, b0, (short)0, s0, false, false);
      s1 = __builtin_amdgcn_wmma_f32_16x16x32_f16(false, aq[ks], false, b1, (short)0, s1, false, false);
    }

    // rel-pos shift add, softcap, causal/pad mask (accumulator layout: col=l16, rows hi16*8+r)
#pragma unroll
    for (int which = 0; which < 2; ++which) {
      int j = j0 + which * 16 + l16;
      int tabs = toff + j;
#pragma unroll
      for (int r = 0; r < 8; ++r) {
        int m = hi16 * 8 + r;
        int i = i0 + m;
        int p = j - i;                       // Tx-XL shift: pos index
        float val = which ? s1[r] : s0[r];
        bool inwin = (p >= 0) && (p < 128);  // == causal window j in [i, i+127]
        if (inwin) val += (float)Tbd[w][m][p];
        val = 50.0f * tanhf(val * 0.02f);
        bool valid = inwin && (tabs >= 0) && (j < NCTX);
        if (valid) valid = (mask[(long)b * TSEQ + tabs] != 0);
        if (!valid) val = -1.0e30f;
        if (which) s1[r] = val; else s0[r] = val;
      }
    }

    // chunk row max (reduce across the 16 lanes owning each row's columns)
    float cmax[8];
#pragma unroll
    for (int r = 0; r < 8; ++r) cmax[r] = fmaxf(s0[r], s1[r]);
#pragma unroll
    for (int off = 8; off >= 1; off >>= 1)
#pragma unroll
      for (int r = 0; r < 8; ++r)
        cmax[r] = fmaxf(cmax[r], __shfl_xor(cmax[r], off, 32));

    float sc[8];
#pragma unroll
    for (int r = 0; r < 8; ++r) {
      float nm = fmaxf(rmax[r], cmax[r]);
      sc[r] = __expf(rmax[r] - nm);
      rmax[r] = nm;
    }
    float csum[8];
#pragma unroll
    for (int r = 0; r < 8; ++r) {
      float e0 = __expf(s0[r] - rmax[r]);
      float e1 = __expf(s1[r] - rmax[r]);
      csum[r] = e0 + e1;
      int m = hi16 * 8 + r;
      Pst[w][m][l16] = (_Float16)e0;
      Pst[w][m][16 + l16] = (_Float16)e1;
    }
#pragma unroll
    for (int off = 8; off >= 1; off >>= 1)
#pragma unroll
      for (int r = 0; r < 8; ++r)
        csum[r] += __shfl_xor(csum[r], off, 32);
#pragma unroll
    for (int r = 0; r < 8; ++r) rsum[r] = rsum[r] * sc[r] + csum[r];
#pragma unroll
    for (int dt = 0; dt < 8; ++dt)
#pragma unroll
      for (int r = 0; r < 8; ++r)
        o[dt][r] *= sc[r];

    __builtin_amdgcn_wave_barrier();   // keep Pst stores before reload (same-wave DS is in-order)
    half16_t ap = load_a_frag(&Pst[w][0][0], l16, PSTRIDE, 0, hi16);

    // O += P @ V from LDS
#pragma unroll
    for (int dt = 0; dt < 8; ++dt) {
      half16_t bv = load_b_frag(&Vst[0][0], dt * 16 + l16, VSTRIDE, 0, hi16);
      o[dt] = __builtin_amdgcn_wmma_f32_16x16x32_f16(false, ap, false, bv,
                                                     (short)0, o[dt], false, false);
    }
    __builtin_amdgcn_wave_barrier();
  }

  // normalize + write [B,T,H,Dh] f32
#pragma unroll
  for (int dt = 0; dt < 8; ++dt) {
#pragma unroll
    for (int r = 0; r < 8; ++r) {
      int m = hi16 * 8 + r;
      int t = nb * CHUNKSZ + i0 + m;
      int d = dt * 16 + l16;
      out[(((long)b * TSEQ + t) * HEADS + h) * HEAD_DIM + d] = o[dt][r] / rsum[r];
    }
  }
}

// ---------------- launcher ----------------

extern "C" void kernel_launch(void* const* d_in, const int* in_sizes, int n_in,
                              void* d_out, int out_size, void* d_ws, size_t ws_size,
                              hipStream_t stream) {
  (void)in_sizes; (void)n_in; (void)out_size; (void)ws_size;
  const float* x    = (const float*)d_in[0];
  const float* Wq   = (const float*)d_in[1];
  const float* Wk   = (const float*)d_in[2];
  const float* Wv   = (const float*)d_in[3];
  const float* Wpos = (const float*)d_in[4];
  const float* pds  = (const float*)d_in[5];
  const unsigned char* mask = (const unsigned char*)d_in[6];  // jnp bool = 1 byte

  char* ws = (char*)d_ws;
  size_t off = 0;
  auto alloc = [&](size_t bytes) { size_t o = off; off += (bytes + 255) & ~(size_t)255; return o; };

  const long MT = (long)BATCH * TSEQ;                         // 16384
  _Float16* xh   = (_Float16*)(ws + alloc((size_t)MT * HIDDEN * 2));
  _Float16* wqt  = (_Float16*)(ws + alloc((size_t)HIDDEN * HIDDEN * 2));
  _Float16* wkt  = (_Float16*)(ws + alloc((size_t)HIDDEN * HIDDEN * 2));
  _Float16* wvt  = (_Float16*)(ws + alloc((size_t)HIDDEN * HIDDEN * 2));
  _Float16* wpt  = (_Float16*)(ws + alloc((size_t)HIDDEN * HIDDEN * 2));
  _Float16* se   = (_Float16*)(ws + alloc((size_t)128 * HIDDEN * 2));
  _Float16* qhb  = (_Float16*)(ws + alloc((size_t)MT * HIDDEN * 2));
  _Float16* khb  = (_Float16*)(ws + alloc((size_t)MT * HIDDEN * 2));
  _Float16* vtb  = (_Float16*)(ws + alloc((size_t)BATCH * HEADS * HEAD_DIM * TSEQ * 2));
  _Float16* posb = (_Float16*)(ws + alloc((size_t)128 * HIDDEN * 2));

  const int thr = 256;
  long nx = MT * HIDDEN;
  conv_x_kernel<<<(unsigned)((nx + thr - 1) / thr), thr, 0, stream>>>(x, xh, nx);
  long wn = (long)HIDDEN * HIDDEN;
  unsigned wgrid = (unsigned)((wn + thr - 1) / thr);
  transpose_w_kernel<<<wgrid, thr, 0, stream>>>(Wq, wqt);
  transpose_w_kernel<<<wgrid, thr, 0, stream>>>(Wk, wkt);
  transpose_w_kernel<<<wgrid, thr, 0, stream>>>(Wv, wvt);
  transpose_w_kernel<<<wgrid, thr, 0, stream>>>(Wpos, wpt);
  sinemb_kernel<<<(128 * HIDDEN + thr - 1) / thr, thr, 0, stream>>>(se);

  unsigned ggrid = (unsigned)((MT / 128) * (HIDDEN / 128));   // 1536
  gemm_f16_kernel<<<ggrid, 256, 0, stream>>>(xh, wqt, qhb, pds, (int)MT, 0);
  gemm_f16_kernel<<<ggrid, 256, 0, stream>>>(xh, wkt, khb, pds, (int)MT, 1);
  gemm_f16_kernel<<<ggrid, 256, 0, stream>>>(xh, wvt, vtb, pds, (int)MT, 2);
  gemm_f16_kernel<<<(HIDDEN / 128), 256, 0, stream>>>(se, wpt, posb, pds, 128, 3);

  attn_kernel<<<BATCH * NBLK * HEADS, 256, 0, stream>>>(qhb, khb, vtb, posb, mask,
                                                        (float*)d_out);
}
